// SoftAgg_88064009437424
// MI455X (gfx1250) — compile-verified, matrix operand order checked by reading, not run on hardware
//
#include <hip/hip_runtime.h>
#include <hip/hip_bf16.h>

// Problem constants (from reference): B=8, N=8192, D=512, G=64
#define B_DIM 8
#define N_DIM 8192
#define D_DIM 512
#define G_DIM 64

#define ECH_W 32          // e-columns per workgroup
#define NBLK_N 1024       // n-rows per workgroup (N / 8 nblks)
#define NT 32             // n-rows per super-tile (2 WMMA row-tiles)
#define NTILES (NBLK_N / NT)

typedef __attribute__((ext_vector_type(16))) __bf16 v16bf;
typedef __attribute__((ext_vector_type(8)))  float  v8f;

__device__ __forceinline__ unsigned short f32_to_bf16_rne(float f) {
  unsigned int u = __float_as_uint(f);
  u += 0x7FFFu + ((u >> 16) & 1u);
  return (unsigned short)(u >> 16);
}
__device__ __forceinline__ unsigned int pack_bf16x2(float lo, float hi) {
  return (unsigned int)f32_to_bf16_rne(lo) | ((unsigned int)f32_to_bf16_rne(hi) << 16);
}

// Low 32 bits of a flat shared pointer == logical LDS byte address (LDS aperture).
__device__ __forceinline__ unsigned lds_addr_of(const void* p) {
  return (unsigned)(uintptr_t)p;
}

// gfx1250 async DMA: global -> LDS, 16 bytes per lane, tracked by ASYNCcnt.
__device__ __forceinline__ void async_copy_b128(unsigned lds_dst, const void* gsrc) {
  asm volatile("global_load_async_to_lds_b128 %0, %1, off"
               :: "v"(lds_dst), "v"(gsrc) : "memory");
}
__device__ __forceinline__ void wait_async0() {
  asm volatile("s_wait_asynccnt 0" ::: "memory");
}

// ---------------- f32 -> bf16 convert (8 elems / thread, exact-size grids) ----------------
__global__ void k_cvt_bf16(const float* __restrict__ src, unsigned short* __restrict__ dst) {
  const size_t gid = (size_t)blockIdx.x * blockDim.x + threadIdx.x;
  const float4 a = ((const float4*)src)[gid * 2 + 0];
  const float4 b = ((const float4*)src)[gid * 2 + 1];
  uint4 r;
  r.x = pack_bf16x2(a.x, a.y);
  r.y = pack_bf16x2(a.z, a.w);
  r.z = pack_bf16x2(b.x, b.y);
  r.w = pack_bf16x2(b.z, b.w);
  ((uint4*)dst)[gid] = r;
}

// ---------------- Fused GEMM1+GEMM2 + online segmented softmax aggregation ----------------
// grid: (nblk=8, ech=16, b=8); block: 256 threads = 8 waves.
// Wave roles: mat = wave&1 (w1 vs w2), esub = (wave>>1)&1 (16 e's), nsub = wave>>2 (16 n's).
__launch_bounds__(256)
__global__ void k_fused(const unsigned short* __restrict__ xb,
                        const unsigned short* __restrict__ w1b,
                        const unsigned short* __restrict__ w2b,
                        const float* __restrict__ b1,
                        const float* __restrict__ b2,
                        const int* __restrict__ ids,
                        float* __restrict__ partials) {
  extern __shared__ char smem[];
  unsigned short* w1s = (unsigned short*)smem;            // 32 x 512 bf16 = 32 KiB
  unsigned short* w2s = w1s + ECH_W * D_DIM;              // 32 KiB
  unsigned short* xs0 = w2s + ECH_W * D_DIM;              // 32 KiB (double buffer A)
  unsigned short* xs1 = xs0 + NT * D_DIM;                 // 32 KiB (double buffer B)
  float* atile = (float*)(xs1 + NT * D_DIM);              // 32 x 32 f32 = 4 KiB
  float* vtile = atile + NT * ECH_W;                      // 4 KiB
  float* smax  = vtile + NT * ECH_W;                      // 64 x 32 f32 = 8 KiB
  float* ssum  = smax + G_DIM * ECH_W;                    // 8 KiB
  float* tsum  = ssum + G_DIM * ECH_W;                    // 8 KiB
  int*   sblk  = (int*)(tsum + G_DIM * ECH_W);            // 1024 ints = 4 KiB

  const int tid  = threadIdx.x;
  const int nblk = blockIdx.x;   // 0..7
  const int ech  = blockIdx.y;   // 0..15
  const int b    = blockIdx.z;   // 0..7
  const int e_base = ech * ECH_W;
  const int n_base = nblk * NBLK_N;
  const unsigned short* xrow_base = xb + ((size_t)b * N_DIM + n_base) * D_DIM;

  // Stage weight slices (bf16) into LDS + full id block + init state.
  {
    const uint4* s1 = (const uint4*)(w1b + (size_t)e_base * D_DIM);
    const uint4* s2 = (const uint4*)(w2b + (size_t)e_base * D_DIM);
    uint4* d1 = (uint4*)w1s;
    uint4* d2 = (uint4*)w2s;
    for (int i = tid; i < ECH_W * D_DIM / 8; i += 256) { d1[i] = s1[i]; d2[i] = s2[i]; }
    ((uint4*)sblk)[tid] = ((const uint4*)(ids + n_base))[tid];  // 1024 ids
  }
  for (int i = tid; i < G_DIM * ECH_W; i += 256) {
    smax[i] = -__builtin_inff(); ssum[i] = 0.f; tsum[i] = 0.f;
  }

  // Prologue: async-stage x tile 0 into xs0 (8 x b128 per thread = 32 KiB).
  {
    const char* src = (const char*)xrow_base;
    const unsigned dst = lds_addr_of(xs0);
#pragma unroll
    for (int j = 0; j < 8; ++j) {
      const unsigned byte = (unsigned)(j * 256 + tid) * 16u;
      async_copy_b128(dst + byte, src + byte);
    }
  }
  wait_async0();
  __syncthreads();

  const int wave = tid >> 5;
  const int lane = tid & 31;
  const int mat  = wave & 1;
  const int esub = (wave >> 1) & 1;
  const int nsub = wave >> 2;
  const unsigned short* wsel = mat ? w2s : w1s;
  const float bias = (mat ? b2 : b1)[e_base + esub * 16 + (lane & 15)];
  const int upd_e    = tid & 31;   // e-column owned in update phase
  const int upd_gmod = tid >> 5;   // group residue class owned (mod 8)

  for (int t = 0; t < NTILES; ++t) {
    const unsigned short* xs = (t & 1) ? xs1 : xs0;

    // Kick off async DMA of tile t+1 into the other buffer (overlaps WMMA sweep),
    // and prefetch tile t+2 into L2 ahead of the async engine.
    if (t + 1 < NTILES) {
      const char* src = (const char*)(xrow_base + (size_t)(t + 1) * NT * D_DIM);
      const unsigned dst = lds_addr_of((t & 1) ? xs0 : xs1);
#pragma unroll
      for (int j = 0; j < 8; ++j) {
        const unsigned byte = (unsigned)(j * 256 + tid) * 16u;
        async_copy_b128(dst + byte, src + byte);
      }
      if (t + 2 < NTILES)
        __builtin_prefetch((const char*)(xrow_base + (size_t)(t + 2) * NT * D_DIM) + tid * 128, 0, 0);
    }

    // 16x16 output tile per wave via 16 chained v_wmma_f32_16x16x32_bf16.
    v8f acc = {0.f, 0.f, 0.f, 0.f, 0.f, 0.f, 0.f, 0.f};
    const int arow = nsub * 16 + (lane & 15);
    const int bcol = esub * 16 + (lane & 15);
    const int kh   = (lane >> 4) * 16;
#pragma unroll
    for (int k0 = 0; k0 < D_DIM; k0 += 32) {
      v16bf afrag = *(const v16bf*)(xs   + (size_t)arow * D_DIM + k0 + kh);
      v16bf bfrag = *(const v16bf*)(wsel + (size_t)bcol * D_DIM + k0 + kh);
      acc = __builtin_amdgcn_wmma_f32_16x16x32_bf16(false, afrag, false, bfrag,
                                                    (short)0, acc, false, false);
    }
    float* tb = mat ? vtile : atile;
    const int rhi = (lane >= 16) ? 8 : 0;
#pragma unroll
    for (int j = 0; j < 8; ++j) {
      tb[(nsub * 16 + rhi + j) * ECH_W + esub * 16 + (lane & 15)] = acc[j] + bias;
    }
    __syncthreads();

    // Online softmax-weighted aggregation across all 256 threads:
    // thread (e, gmod) applies rows whose group satisfies g%8 == gmod.
    // Race-free: state slot g*32+e is owned by exactly one thread.
    {
      const int* sid = sblk + t * NT;
      for (int i = 0; i < NT; ++i) {
        const int g = sid[i];
        if ((g & 7) == upd_gmod) {
          const float a  = atile[i * ECH_W + upd_e];
          const float vv = vtile[i * ECH_W + upd_e];
          const int idx = g * ECH_W + upd_e;
          const float mo = smax[idx];
          const float mn = fmaxf(mo, a);
          const float sc = __expf(mo - mn);   // exp(-inf)=0 handles first touch
          const float w  = __expf(a - mn);
          smax[idx] = mn;
          ssum[idx] = ssum[idx] * sc + w;
          tsum[idx] = tsum[idx] * sc + w * vv;
        }
      }
    }
    // Close the pipeline stage: next tile's LDS buffer must be fully written.
    wait_async0();
    __syncthreads();
  }

  // Dump partial state: [wgid][3][G*ECH_W]
  const size_t wgid = ((size_t)b * 16 + ech) * 8 + nblk;
  float* outp = partials + wgid * 3 * (G_DIM * ECH_W);
  for (int i = tid; i < G_DIM * ECH_W; i += 256) {
    outp[i]                     = smax[i];
    outp[G_DIM * ECH_W + i]     = ssum[i];
    outp[2 * G_DIM * ECH_W + i] = tsum[i];
  }
}

// ---------------- Combine partial softmax states -> y (bf16) ----------------
__global__ void k_combine(const float* __restrict__ partials,
                          unsigned short* __restrict__ yb) {
  const int ech = blockIdx.x;  // 0..15
  const int b   = blockIdx.y;  // 0..7
  const int tid = threadIdx.x;
  const int SL  = G_DIM * ECH_W;  // 2048
  for (int s = tid; s < SL; s += 256) {
    float m = -__builtin_inff(), ss = 0.f, tt = 0.f;
    for (int p = 0; p < 8; ++p) {
      const float* base = partials + ((((size_t)b * 16 + ech) * 8 + p) * 3 * SL);
      const float pm = base[s];
      const float ps = base[SL + s];
      const float pt = base[2 * SL + s];
      const float nm = fmaxf(m, pm);
      const float e1 = (m  == -__builtin_inff()) ? 0.f : __expf(m - nm);
      const float e2 = (pm == -__builtin_inff()) ? 0.f : __expf(pm - nm);
      ss = ss * e1 + ps * e2;
      tt = tt * e1 + pt * e2;
      m = nm;
    }
    const float y = (ss > 0.f) ? (tt / ss) : 0.f;
    const int g = s / ECH_W;
    const int e = s % ECH_W;
    yb[((size_t)b * G_DIM + g) * D_DIM + ech * ECH_W + e] = f32_to_bf16_rne(y);
  }
}

// ---------------- GEMM3: out_small[m,o] = y[m,:] @ w3[o,:]ᵀ + b3[o], M=512 ----------------
__launch_bounds__(256)
__global__ void k_gemm3(const unsigned short* __restrict__ yb,
                        const unsigned short* __restrict__ w3b,
                        const float* __restrict__ b3,
                        float* __restrict__ out_small) {
  const int tid  = threadIdx.x;
  const int wave = tid >> 5, lane = tid & 31;
  const int tileId = blockIdx.x * 8 + wave;   // 1024 tiles of 16x16
  const int m0 = (tileId >> 5) * 16;
  const int o0 = (tileId & 31) * 16;
  const int kh = (lane >> 4) * 16;
  const int arow = m0 + (lane & 15);
  const int bcol = o0 + (lane & 15);
  v8f acc = {0.f, 0.f, 0.f, 0.f, 0.f, 0.f, 0.f, 0.f};
#pragma unroll
  for (int k0 = 0; k0 < D_DIM; k0 += 32) {
    v16bf af = *(const v16bf*)(yb  + (size_t)arow * D_DIM + k0 + kh);
    v16bf bf = *(const v16bf*)(w3b + (size_t)bcol * D_DIM + k0 + kh);
    acc = __builtin_amdgcn_wmma_f32_16x16x32_bf16(false, af, false, bf,
                                                  (short)0, acc, false, false);
  }
  const float bias = b3[o0 + (lane & 15)];
  const int rhi = (lane >= 16) ? 8 : 0;
#pragma unroll
  for (int j = 0; j < 8; ++j) {
    out_small[(size_t)(m0 + rhi + j) * D_DIM + o0 + (lane & 15)] = acc[j] + bias;
  }
}

// ---------------- Expand: out[b,n,:] = out_small[b, ids[n], :]  (float4) ----------------
__global__ void k_expand(const float* __restrict__ out_small,
                         const int* __restrict__ ids,
                         float* __restrict__ out) {
  const size_t q = (size_t)blockIdx.x * blockDim.x + threadIdx.x;  // quad index
  const int o4 = (int)(q & 127);           // D/4 = 128
  const size_t row = q >> 7;               // b*N + n
  const int n = (int)(row & (N_DIM - 1));
  const int b = (int)(row >> 13);
  const int g = ids[n];
  ((float4*)out)[q] = ((const float4*)out_small)[((size_t)b * G_DIM + g) * 128 + o4];
}

extern "C" void kernel_launch(void* const* d_in, const int* in_sizes, int n_in,
                              void* d_out, int out_size, void* d_ws, size_t ws_size,
                              hipStream_t stream) {
  const float* x   = (const float*)d_in[0];
  const int*   ids = (const int*)d_in[1];
  const float* w1  = (const float*)d_in[2];
  const float* b1  = (const float*)d_in[3];
  const float* w2  = (const float*)d_in[4];
  const float* b2  = (const float*)d_in[5];
  const float* w3  = (const float*)d_in[6];
  const float* b3  = (const float*)d_in[7];

  char* ws = (char*)d_ws;
  unsigned short* xb  = (unsigned short*)(ws);              // 64 MiB   x bf16
  unsigned short* w1b = (unsigned short*)(ws + 67108864);   // 512 KiB
  unsigned short* w2b = (unsigned short*)(ws + 67633152);   // 512 KiB
  unsigned short* w3b = (unsigned short*)(ws + 68157440);   // 512 KiB
  float* partials     = (float*)(ws + 68681728);            // 24 MiB   (1024 wg x 3 x 2048 f32)
  unsigned short* yb  = (unsigned short*)(ws + 93847552);   // 512 KiB  y bf16 [512][512]
  float* out_small    = (float*)(ws + 94371840);            // 1 MiB    [512][512] f32

  // 1) downconvert x and weights to bf16 (exact-size grids, 8 elems/thread)
  k_cvt_bf16<<<16384, 256, 0, stream>>>(x,  xb);   // 33,554,432 elems
  k_cvt_bf16<<<128,   256, 0, stream>>>(w1, w1b);  // 262,144 elems
  k_cvt_bf16<<<128,   256, 0, stream>>>(w2, w2b);
  k_cvt_bf16<<<128,   256, 0, stream>>>(w3, w3b);

  // 2) fused WMMA GEMM1/GEMM2 + online segmented softmax aggregation
  const size_t smem = (size_t)(ECH_W * D_DIM * 2) * 2     // w1s + w2s
                    + (size_t)(NT * D_DIM * 2) * 2        // xs0 + xs1 (double buffer)
                    + (size_t)(NT * ECH_W * 4) * 2        // atile + vtile
                    + (size_t)(G_DIM * ECH_W * 4) * 3     // smax/ssum/tsum
                    + (size_t)(NBLK_N * 4);               // id block
  k_fused<<<dim3(8, 16, 8), 256, smem, stream>>>(xb, w1b, w2b, b1, b2, ids, partials);

  // 3) merge partial online-softmax states -> y (bf16)
  k_combine<<<dim3(16, 8), 256, 0, stream>>>(partials, yb);

  // 4) small WMMA GEMM3 + bias
  k_gemm3<<<128, 256, 0, stream>>>(yb, w3b, b3, out_small);

  // 5) gather-expand to [B,N,D]
  k_expand<<<32768, 256, 0, stream>>>(out_small, ids, (float*)d_out);
}
